// HexagonalMultiHeadAttention_63488206569714
// MI455X (gfx1250) — matrix operand order repaired and checked by reading, not compiled, not run on hardware
//
#include <hip/hip_runtime.h>
#include <hip/hip_bf16.h>

#define B_   2
#define S_   2048
#define D_   1024
#define DH_  64
#define H_   19
#define HD_  1216   // H_*DH_

typedef __attribute__((ext_vector_type(16))) __bf16 v16bf;
typedef __attribute__((ext_vector_type(8)))  __bf16 v8bf;
typedef __attribute__((ext_vector_type(8)))  float  v8f;

static __device__ __forceinline__ unsigned short f2bf(float f){
  unsigned int u = __float_as_uint(f);
  unsigned int r = u + 0x7FFFu + ((u >> 16) & 1u);   // round-to-nearest-even
  return (unsigned short)(r >> 16);
}
static __device__ __forceinline__ float bf2f(unsigned short s){
  return __uint_as_float(((unsigned int)s) << 16);
}
static __device__ __forceinline__ v8f zero8(){
  v8f z = {0.f,0.f,0.f,0.f,0.f,0.f,0.f,0.f}; return z;
}
static __device__ __forceinline__ v8f wmma_bf16(v16bf a, v16bf b, v8f c){
  return __builtin_amdgcn_wmma_f32_16x16x32_bf16(false, a, false, b, (short)0, c, false, false);
}
static __device__ __forceinline__ v16bf combine(v8bf lo, v8bf hi){
  return __builtin_shufflevector(lo, hi, 0,1,2,3,4,5,6,7,8,9,10,11,12,13,14,15);
}

// A fragment (16x32): lane holds row m0+lane%16; one contiguous 16-elem run.
static __device__ __forceinline__ v16bf frag_a(const unsigned short* base, int ld, int m0, int k0, int lane){
  const unsigned short* p = base + (m0 + (lane & 15))*ld + k0 + ((lane >> 4) * 16);
  v8bf lo = *(const v8bf*)p;
  v8bf hi = *(const v8bf*)(p + 8);
  return combine(lo, hi);
}
// B fragment (32x16) from contraction-major [n][k] storage: one contiguous 16-elem run.
static __device__ __forceinline__ v16bf frag_brow(const unsigned short* base, int ld, int n0, int k0, int lane){
  const unsigned short* p = base + (n0 + (lane & 15))*ld + k0 + ((lane >> 4) * 16);
  v8bf lo = *(const v8bf*)p;
  v8bf hi = *(const v8bf*)(p + 8);
  return combine(lo, hi);
}
// B fragment from pre-swizzled fragment-major storage [tile][lane][16]: 32B contiguous.
static __device__ __forceinline__ v16bf frag_pack(const unsigned short* base, int nt, int lane){
  return *(const v16bf*)(base + nt*512 + lane*16);
}

static __device__ __forceinline__ float red_max16(float v){
  #pragma unroll
  for (int off = 8; off >= 1; off >>= 1) v = fmaxf(v, __shfl_xor(v, off, 32));
  return v;
}
static __device__ __forceinline__ float red_sum16(float v){
  #pragma unroll
  for (int off = 8; off >= 1; off >>= 1) v += __shfl_xor(v, off, 32);
  return v;
}

// ---------------- stage 0: coordinate-bias MLP (tiny) ----------------
__global__ void bias_kernel(const float* __restrict__ coords, const float* __restrict__ wb1,
                            const float* __restrict__ bb1, const float* __restrict__ wb2,
                            const float* __restrict__ bb2, float* __restrict__ bias){
  const int h = blockIdx.x, e = threadIdx.x;
  const float c0 = coords[2*h], c1 = coords[2*h+1];
  float acc = bb2[e];
  for (int d = 0; d < D_; ++d){
    float hv = fmaxf(c0*wb1[d] + c1*wb1[D_+d] + bb1[d], 0.0f);
    acc += hv * wb2[d*DH_ + e];
  }
  bias[h*DH_ + e] = acc;
}

// ---------------- stage 1: conversions / weight swizzles ----------------
__global__ void cvt_kernel(const float* __restrict__ in, unsigned short* __restrict__ out, int n){
  int i = blockIdx.x*blockDim.x + threadIdx.x;
  if (i < n) out[i] = f2bf(in[i]);
}
__global__ void swz_head_kernel(const float* __restrict__ in, unsigned short* __restrict__ out){
  int idx = blockIdx.x*blockDim.x + threadIdx.x;
  int h = idx >> 16, rem = idx & 65535;
  int k = rem >> 6, n = rem & 63;
  int c = k >> 5, half = (k >> 4) & 1, j = k & 15;
  int lane = half*16 + (n & 15), nt = n >> 4;
  out[h*65536 + ((c*4 + nt)*32 + lane)*16 + j] = f2bf(in[idx]);
}
__global__ void swz_wo_kernel(const float* __restrict__ in, unsigned short* __restrict__ out){
  int idx = blockIdx.x*blockDim.x + threadIdx.x;
  int k = idx >> 10, n = idx & 1023;
  int c = k >> 5, half = (k >> 4) & 1, j = k & 15;
  int lane = half*16 + (n & 15), T = n >> 4;
  out[((c*64 + T)*32 + lane)*16 + j] = f2bf(in[idx]);
}

// ---------------- stage 2: fused QKV projection GEMM (double-buffered) ----------------
__global__ __launch_bounds__(256) void qkv_kernel(
    const unsigned short* __restrict__ xb,
    const unsigned short* __restrict__ wqb,
    const unsigned short* __restrict__ wkb,
    const unsigned short* __restrict__ wvb,
    const float* __restrict__ bias,
    unsigned short* __restrict__ qb,
    unsigned short* __restrict__ kb,
    unsigned short* __restrict__ vbT){
  __shared__ __align__(16) unsigned short Xs[2][64][40];
  __shared__ __align__(32) unsigned short Wf[2][3][2048];
  const int tid = threadIdx.x, lane = tid & 31, wv = tid >> 5;
  const int h = blockIdx.y;
  const int row0 = blockIdx.x * 64;
  const int mi = wv & 3, nq = wv >> 2;
  const int r = tid >> 2, q = tid & 3;
  const unsigned short* xrow = xb + (row0 + r)*D_ + q*8;
  const int hbase = h*65536;

  v8f acc[3][2];
  #pragma unroll
  for (int m = 0; m < 3; ++m){ acc[m][0] = zero8(); acc[m][1] = zero8(); }

  auto fill = [&](int buf, int kc){
    *(v8bf*)&Xs[buf][r][q*8] = *(const v8bf*)(xrow + kc);
    const int co = hbase + (kc >> 5) * 2048 + tid*8;
    *(v8bf*)(Wf[buf][0] + tid*8) = *(const v8bf*)(wqb + co);
    *(v8bf*)(Wf[buf][1] + tid*8) = *(const v8bf*)(wkb + co);
    *(v8bf*)(Wf[buf][2] + tid*8) = *(const v8bf*)(wvb + co);
    if (kc + 64 < D_) __builtin_prefetch(xrow + kc + 64, 0, 3);
  };

  fill(0, 0);
  __syncthreads();
  for (int it = 0; it < D_/32; ++it){
    const int cur = it & 1;
    if (it + 1 < D_/32) fill(1 - cur, (it + 1)*32);
    v16bf a = frag_a(&Xs[cur][0][0], 40, mi*16, 0, lane);
    v16bf bfr[6];
    #pragma unroll
    for (int mat = 0; mat < 3; ++mat)
      #pragma unroll
      for (int t = 0; t < 2; ++t)
        bfr[mat*2 + t] = frag_pack(Wf[cur][mat], nq*2 + t, lane);
    #pragma unroll
    for (int mat = 0; mat < 3; ++mat)
      #pragma unroll
      for (int t = 0; t < 2; ++t)
        acc[mat][t] = wmma_bf16(a, bfr[mat*2 + t], acc[mat][t]);
    __syncthreads();
  }
  const int half = lane >> 4, nin = lane & 15;
  #pragma unroll
  for (int t = 0; t < 2; ++t){
    const int e = nq*32 + t*16 + nin;
    const float bq = bias[h*DH_ + e];
    #pragma unroll
    for (int j = 0; j < 8; ++j){
      int row = row0 + mi*16 + j + 8*half;
      int b = row >> 11, s = row & (S_-1);
      int hb = b*H_ + h;
      qb[(hb*S_ + s)*DH_ + e]   = f2bf((acc[0][t][j] + bq) * 0.125f); // fold 1/sqrt(DH)
      kb[(hb*S_ + s)*DH_ + e]   = f2bf(acc[1][t][j]);
      vbT[hb*DH_*S_ + e*S_ + s] = f2bf(acc[2][t][j]);                  // transposed for PV
    }
  }
}

// ---------------- stage 3: flash attention (double-buffered K/V) ----------------
__global__ __launch_bounds__(256) void attn_kernel(
    const unsigned short* __restrict__ qb,
    const unsigned short* __restrict__ kb,
    const unsigned short* __restrict__ vbT,
    float* __restrict__ heads){
  __shared__ __align__(16) unsigned short Ks[2][64][72];   // [t][e]
  __shared__ __align__(16) unsigned short Vt[2][64][72];   // [e][t]
  __shared__ __align__(16) unsigned short Ps[8][16][72];
  const int tid = threadIdx.x, lane = tid & 31, wv = tid >> 5;
  const int h = blockIdx.y, b = blockIdx.z;
  const int hb = b*H_ + h;
  const int base = hb * S_ * DH_;
  const int m0 = blockIdx.x*128 + wv*16;
  v16bf qa0 = frag_a(qb + base, DH_, m0, 0, lane);
  v16bf qa1 = frag_a(qb + base, DH_, m0, 32, lane);
  v8f oacc[4]; float mrun[8], lrun[8];
  #pragma unroll
  for (int nt = 0; nt < 4; ++nt) oacc[nt] = zero8();
  #pragma unroll
  for (int j = 0; j < 8; ++j){ mrun[j] = -1.0e30f; lrun[j] = 0.0f; }

  const int rr = tid >> 3, qq8 = (tid & 7) * 8;
  auto fillKV = [&](int buf, int t0){
    #pragma unroll
    for (int i = 0; i < 2; ++i){
      int rrr = rr + i*32;
      *(v8bf*)&Ks[buf][rrr][qq8] = *(const v8bf*)(kb  + base + (t0 + rrr)*DH_ + qq8);
      *(v8bf*)&Vt[buf][rrr][qq8] = *(const v8bf*)(vbT + base + rrr*S_ + t0 + qq8);
    }
  };

  fillKV(0, 0);
  __syncthreads();
  for (int it = 0; it < S_/64; ++it){
    const int cur = it & 1;
    if (it + 1 < S_/64) fillKV(1 - cur, (it + 1)*64);

    // scores: S[m][t] = sum_e q[m][e]*k[t][e]
    v16bf kf[8];
    #pragma unroll
    for (int nt = 0; nt < 4; ++nt){
      kf[2*nt]   = frag_brow(&Ks[cur][0][0], 72, nt*16, 0,  lane);
      kf[2*nt+1] = frag_brow(&Ks[cur][0][0], 72, nt*16, 32, lane);
    }
    v8f sacc[4];
    #pragma unroll
    for (int nt = 0; nt < 4; ++nt){
      v8f s = zero8();
      s = wmma_bf16(qa0, kf[2*nt],   s);
      s = wmma_bf16(qa1, kf[2*nt+1], s);
      sacc[nt] = s;
    }
    const int mrow = 8*(lane >> 4);
    #pragma unroll
    for (int j = 0; j < 8; ++j){
      float rmax = fmaxf(fmaxf(sacc[0][j], sacc[1][j]), fmaxf(sacc[2][j], sacc[3][j]));
      rmax = red_max16(rmax);
      float mnew = fmaxf(mrun[j], rmax);
      float corr = __expf(mrun[j] - mnew);
      float pv[4]; float rs = 0.0f;
      #pragma unroll
      for (int nt = 0; nt < 4; ++nt){ pv[nt] = __expf(sacc[nt][j] - mnew); rs += pv[nt]; }
      rs = red_sum16(rs);
      lrun[j] = lrun[j]*corr + rs;
      mrun[j] = mnew;
      #pragma unroll
      for (int nt = 0; nt < 4; ++nt) oacc[nt][j] *= corr;
      #pragma unroll
      for (int nt = 0; nt < 4; ++nt)
        Ps[wv][j + mrow][nt*16 + (lane & 15)] = f2bf(pv[nt]);
    }
    v16bf pa0 = frag_a(&Ps[wv][0][0], 72, 0, 0, lane);
    v16bf pa1 = frag_a(&Ps[wv][0][0], 72, 0, 32, lane);
    v16bf vf[8];
    #pragma unroll
    for (int nt = 0; nt < 4; ++nt){
      vf[2*nt]   = frag_brow(&Vt[cur][0][0], 72, nt*16, 0,  lane);
      vf[2*nt+1] = frag_brow(&Vt[cur][0][0], 72, nt*16, 32, lane);
    }
    #pragma unroll
    for (int nt = 0; nt < 4; ++nt){
      oacc[nt] = wmma_bf16(pa0, vf[2*nt],   oacc[nt]);
      oacc[nt] = wmma_bf16(pa1, vf[2*nt+1], oacc[nt]);
    }
    __syncthreads();
  }
  const int half = lane >> 4, nin = lane & 15;
  #pragma unroll
  for (int j = 0; j < 8; ++j){
    float inv = 1.0f / lrun[j];
    int s = m0 + j + 8*half;
    #pragma unroll
    for (int nt = 0; nt < 4; ++nt)
      heads[base + s*DH_ + nt*16 + nin] = oacc[nt][j] * inv;
  }
}

// ---------------- stage 4: msg = relu(heads @ w_mix) ----------------
__global__ __launch_bounds__(256) void mix_kernel(
    const float* __restrict__ heads,
    const float* __restrict__ w_mix,
    unsigned short* __restrict__ msg){
  __shared__ __align__(16) unsigned short Hs[128][40];
  __shared__ __align__(16) unsigned short Wm[64][72];   // transposed [n][k]
  const int tid = threadIdx.x, lane = tid & 31, wv = tid >> 5;
  const int row0 = blockIdx.x * 128;
  for (int idx = tid; idx < 128*32; idx += 256){
    int rr = idx >> 5, e2 = (idx & 31) * 2;
    const float* hp = heads + (row0 + rr)*DH_ + e2;
    unsigned int pk = (unsigned int)f2bf(hp[0]) | ((unsigned int)f2bf(hp[1]) << 16);
    *(unsigned int*)&Hs[rr][e2] = pk;
  }
  for (int idx = tid; idx < 64*64; idx += 256){
    int k = idx >> 6, n = idx & 63;
    Wm[n][k] = f2bf(w_mix[k*DH_ + n]);
  }
  __syncthreads();
  v16bf a0 = frag_a(&Hs[0][0], 40, wv*16, 0, lane);
  v16bf a1 = frag_a(&Hs[0][0], 40, wv*16, 32, lane);
  const int half = lane >> 4, nin = lane & 15;
  #pragma unroll
  for (int nt = 0; nt < 4; ++nt){
    v8f acc = zero8();
    acc = wmma_bf16(a0, frag_brow(&Wm[0][0], 72, nt*16, 0,  lane), acc);
    acc = wmma_bf16(a1, frag_brow(&Wm[0][0], 72, nt*16, 32, lane), acc);
    #pragma unroll
    for (int j = 0; j < 8; ++j){
      int rr = row0 + wv*16 + j + 8*half;
      msg[rr*DH_ + nt*16 + nin] = f2bf(fmaxf(acc[j], 0.0f));
    }
  }
}

// ---------------- stage 5: adjacency mix + concat(transposed) ----------------
__global__ __launch_bounds__(256) void concat_kernel(
    const float* __restrict__ heads,
    const unsigned short* __restrict__ msg,
    const float* __restrict__ adj,
    unsigned short* __restrict__ cc){
  __shared__ unsigned short Ms[19][1024];
  __shared__ float Aj[19*19];
  const int tid = threadIdx.x;
  const int s0 = blockIdx.x * 16, b = blockIdx.y;
  for (int idx = tid; idx < 19*19; idx += 256) Aj[idx] = adj[idx];
  for (int idx = tid; idx < 19*16*64; idx += 256){
    int hh = idx >> 10, rem = idx & 1023;
    int rr = rem >> 6, e = rem & 63;
    Ms[hh][rem] = msg[((b*H_ + hh)*S_ + s0 + rr)*DH_ + e];
  }
  __syncthreads();
  for (int idx = tid; idx < 19*16*64; idx += 256){
    int hh = idx >> 10, rem = idx & 1023;
    int rr = rem >> 6, e = rem & 63;
    float acc = heads[((b*H_ + hh)*S_ + s0 + rr)*DH_ + e];
    for (int g = 0; g < H_; ++g)
      acc += Aj[hh*H_ + g] * bf2f(Ms[g][rem]);
    cc[(b*S_ + s0 + rr)*HD_ + hh*DH_ + e] = f2bf(acc);
  }
}

// ---------------- stage 6: output projection GEMM (double-buffered) ----------------
__global__ __launch_bounds__(256) void out_kernel(
    const unsigned short* __restrict__ cc,
    const unsigned short* __restrict__ wob,
    float* __restrict__ out){
  __shared__ __align__(16) unsigned short As[2][64][40];
  __shared__ __align__(32) unsigned short Bf[2][2048];
  const int tid = threadIdx.x, lane = tid & 31, wv = tid >> 5;
  const int row0 = blockIdx.x * 64, col0 = blockIdx.y * 64;
  const int mi = wv & 3, nq = wv >> 2;
  v8f acc0 = zero8(), acc1 = zero8();
  const int r = tid >> 2, q = tid & 3;
  const unsigned short* crow = cc + (row0 + r)*HD_ + q*8;
  const unsigned short* wcol = wob + (col0 >> 4)*512 + tid*8;

  auto fill = [&](int buf, int kc){
    *(v8bf*)&As[buf][r][q*8] = *(const v8bf*)(crow + kc);
    *(v8bf*)(Bf[buf] + tid*8) = *(const v8bf*)(wcol + (kc >> 5)*32768);
    if (kc + 64 < HD_) __builtin_prefetch(crow + kc + 64, 0, 3);
  };

  fill(0, 0);
  __syncthreads();
  for (int it = 0; it < HD_/32; ++it){
    const int cur = it & 1;
    if (it + 1 < HD_/32) fill(1 - cur, (it + 1)*32);
    v16bf a  = frag_a(&As[cur][0][0], 40, mi*16, 0, lane);
    v16bf b0 = frag_pack(Bf[cur], nq*2,     lane);
    v16bf b1 = frag_pack(Bf[cur], nq*2 + 1, lane);
    acc0 = wmma_bf16(a, b0, acc0);
    acc1 = wmma_bf16(a, b1, acc1);
    __syncthreads();
  }
  const int half = lane >> 4, nin = lane & 15;
  #pragma unroll
  for (int j = 0; j < 8; ++j){
    int row = row0 + mi*16 + j + 8*half;
    out[row*D_ + col0 + nq*32 + nin]      = acc0[j];
    out[row*D_ + col0 + nq*32 + 16 + nin] = acc1[j];
  }
}

extern "C" void kernel_launch(void* const* d_in, const int* in_sizes, int n_in,
                              void* d_out, int out_size, void* d_ws, size_t ws_size,
                              hipStream_t stream){
  const float* x      = (const float*)d_in[0];
  const float* coords = (const float*)d_in[1];
  const float* w_q    = (const float*)d_in[2];
  const float* w_k    = (const float*)d_in[3];
  const float* w_v    = (const float*)d_in[4];
  const float* wb1    = (const float*)d_in[5];
  const float* bb1    = (const float*)d_in[6];
  const float* wb2    = (const float*)d_in[7];
  const float* bb2    = (const float*)d_in[8];
  const float* w_mix  = (const float*)d_in[9];
  const float* adj    = (const float*)d_in[10];
  const float* w_o    = (const float*)d_in[11];
  float* out = (float*)d_out;

  char* w = (char*)d_ws;
  auto alloc = [&](size_t bytes){ void* p = (void*)w; w += (bytes + 255) & ~(size_t)255; return p; };
  float*          bias = (float*)         alloc((size_t)H_*DH_*4);
  unsigned short* xb   = (unsigned short*)alloc((size_t)B_*S_*D_*2);
  unsigned short* wqb  = (unsigned short*)alloc((size_t)H_*D_*DH_*2);
  unsigned short* wkb  = (unsigned short*)alloc((size_t)H_*D_*DH_*2);
  unsigned short* wvb  = (unsigned short*)alloc((size_t)H_*D_*DH_*2);
  unsigned short* wob  = (unsigned short*)alloc((size_t)HD_*D_*2);
  unsigned short* qb   = (unsigned short*)alloc((size_t)B_*H_*S_*DH_*2);
  unsigned short* kb   = (unsigned short*)alloc((size_t)B_*H_*S_*DH_*2);
  unsigned short* vbT  = (unsigned short*)alloc((size_t)B_*H_*S_*DH_*2);
  float*          hds  = (float*)         alloc((size_t)B_*H_*S_*DH_*4);
  unsigned short* msg  = (unsigned short*)alloc((size_t)B_*H_*S_*DH_*2);
  unsigned short* cc   = (unsigned short*)alloc((size_t)B_*S_*HD_*2);

  bias_kernel<<<dim3(H_), dim3(DH_), 0, stream>>>(coords, wb1, bb1, wb2, bb2, bias);
  const int nx = B_*S_*D_;
  cvt_kernel<<<(nx+255)/256, 256, 0, stream>>>(x, xb, nx);
  const int nw = H_*D_*DH_;
  swz_head_kernel<<<nw/256, 256, 0, stream>>>(w_q, wqb);
  swz_head_kernel<<<nw/256, 256, 0, stream>>>(w_k, wkb);
  swz_head_kernel<<<nw/256, 256, 0, stream>>>(w_v, wvb);
  swz_wo_kernel<<<(HD_*D_)/256, 256, 0, stream>>>(w_o, wob);

  qkv_kernel<<<dim3((B_*S_)/64, H_), 256, 0, stream>>>(xb, wqb, wkb, wvb, bias, qb, kb, vbT);
  attn_kernel<<<dim3(S_/128, H_, B_), 256, 0, stream>>>(qb, kb, vbT, hds);
  mix_kernel<<<dim3((B_*H_*S_)/128), 256, 0, stream>>>(hds, w_mix, msg);
  concat_kernel<<<dim3(S_/16, B_), 256, 0, stream>>>(hds, msg, adj, cc);
  out_kernel<<<dim3((B_*S_)/64, D_/64), 256, 0, stream>>>(cc, wob, out);
}
